// LinearSparse_84834194030971
// MI455X (gfx1250) — compile-verified, matrix operand order checked by reading, not run on hardware
//
#include <hip/hip_runtime.h>

// Problem dims (from reference): B=8192, IN=1024, OUT=1024, fp32.
constexpr int BATCH = 8192;
constexpr int KDIM  = 1024;   // IN  (reduction)
constexpr int NDIM  = 1024;   // OUT

constexpr int BM = 128, BN = 128, BK = 32;
constexpr int THREADS = 256;              // 8 wave32
constexpr int KSTEPS  = KDIM / BK;        // 32
constexpr int LDSS    = BK + 8;           // padded LDS row stride (bf16 elems, 80B)

typedef __attribute__((ext_vector_type(16))) __bf16 v16bf;
typedef __attribute__((ext_vector_type(8)))  __bf16 v8bf;
typedef __attribute__((ext_vector_type(4)))  __bf16 v4bf;
typedef __attribute__((ext_vector_type(8)))  float  v8f;
typedef __attribute__((ext_vector_type(4)))  float  v4f;
typedef __attribute__((ext_vector_type(4)))  int    v4i;

#ifndef __has_builtin
#define __has_builtin(x) 0
#endif

#if __has_builtin(__builtin_amdgcn_global_load_async_to_lds_b128)
#define HAVE_ASYNC 1
#else
#define HAVE_ASYNC 0
#endif

#if HAVE_ASYNC
#  if __has_builtin(__builtin_amdgcn_s_wait_asynccnt)
#    define WAIT_ASYNC(n) __builtin_amdgcn_s_wait_asynccnt(n)
#  else
#    define WAIT_ASYNC(n) asm volatile("s_wait_asynccnt " #n ::: "memory")
#  endif
#else
#  define WAIT_ASYNC(n)
#endif

__device__ __forceinline__ void async_copy_b128(const __bf16* g, __bf16* l) {
#if HAVE_ASYNC
    __builtin_amdgcn_global_load_async_to_lds_b128(
        (__attribute__((address_space(1))) v4i*)g,
        (__attribute__((address_space(3))) v4i*)l, 0, 0);
#else
    *(v8bf*)l = *(const v8bf*)g;   // global_load_b128 + ds_store_b128 fallback
#endif
}

union Frag16 { v16bf v; v8bf h[2]; };

// A/B bf16 16x32 fragment per ISA layout: lane holds row (M or N) = lane%16,
// half-wave h=lane/16 selects K sub-chunks [8h..8h+7] and [16+8h..16+8h+7].
__device__ __forceinline__ v16bf frag_from_lds(const __bf16* row, int kh) {
    Frag16 f;
    f.h[0] = *(const v8bf*)(row + 8 * kh);
    f.h[1] = *(const v8bf*)(row + 16 + 8 * kh);
    return f.v;
}

// ---------------------------------------------------------------------------
// Kernel 1: fp32 -> (bf16 hi, bf16 residual lo) plane split (bandwidth bound)
// ---------------------------------------------------------------------------
__global__ __launch_bounds__(THREADS, 1)
void split_bf16(const float* __restrict__ X, const float* __restrict__ W,
                __bf16* __restrict__ xhi, __bf16* __restrict__ xlo,
                __bf16* __restrict__ whi, __bf16* __restrict__ wlo)
{
    const size_t NX4 = (size_t)BATCH * KDIM / 4;   // 2M float4 groups
    const size_t NW4 = (size_t)NDIM  * KDIM / 4;   // 256K float4 groups
    const size_t i = (size_t)blockIdx.x * blockDim.x + threadIdx.x;

    const float* src;
    __bf16 *dh, *dl;
    size_t j;
    if (i < NX4)            { src = X; dh = xhi; dl = xlo; j = i; }
    else if (i < NX4 + NW4) { src = W; dh = whi; dl = wlo; j = i - NX4; }
    else return;

    const v4f v = *(const v4f*)(src + j * 4);
    v4bf h, l;
    #pragma unroll
    for (int e = 0; e < 4; ++e) {
        const __bf16 hb = (__bf16)v[e];
        h[e] = hb;
        l[e] = (__bf16)(v[e] - (float)hb);
    }
    *(v4bf*)(dh + j * 4) = h;
    *(v4bf*)(dl + j * 4) = l;
}

// ---------------------------------------------------------------------------
// Kernel 2: WMMA GEMM on bf16 planes; async global->LDS double buffering.
// out = (X.Wt) * mask + bias, 3-term bf16 split product for ~fp32 accuracy.
// ---------------------------------------------------------------------------
__global__ __launch_bounds__(THREADS, 1)
void gemm_bf16x3_async(const __bf16* __restrict__ xhi, const __bf16* __restrict__ xlo,
                       const __bf16* __restrict__ whi, const __bf16* __restrict__ wlo,
                       const float* __restrict__ MSK, const float* __restrict__ BIAS,
                       float* __restrict__ OUT)
{
    extern __shared__ __bf16 smem[];   // [2 bufs][4 arrays][BM][LDSS] = 80KB

    const int tid  = threadIdx.x;
    const int lane = tid & 31;
    const int wv   = tid >> 5;
    const int wm   = wv & 3;      // 4 slices of 32 rows
    const int wn   = wv >> 2;     // 2 slices of 64 cols
    const int l16  = lane & 15;
    const int kh   = lane >> 4;

    const int bm = blockIdx.y * BM;
    const int bn = blockIdx.x * BN;

    // async staging assignment: wave -> plane (0=Xhi 1=Xlo 2=Whi 3=Wlo), half
    const int arr   = wv >> 1;                 // 0..3
    const int rhalf = (wv & 1) * 64;           // rows 0-63 / 64-127
    const int rl    = lane >> 2;               // 0..7 row within 8-row block
    const int cl    = (lane & 3) * 8;          // 0,8,16,24 (bf16 elems, 16B)
    const __bf16* gplane = (arr == 0) ? xhi : (arr == 1) ? xlo
                         : (arr == 2) ? whi : wlo;
    const int grow0 = (arr < 2) ? bm : bn;

    auto slab = [&](int buf, int a) -> __bf16* {
        return smem + ((size_t)(buf * 4 + a) * BM * LDSS);
    };
    auto issue = [&](int kt, int buf) {
        __bf16* dst0 = slab(buf, arr);
        #pragma unroll
        for (int i = 0; i < 8; ++i) {
            const int row = rhalf + i * 8 + rl;
            const size_t goff = (size_t)(grow0 + row) * KDIM + kt * BK + cl;
            async_copy_b128(gplane + goff, dst0 + row * LDSS + cl);
        }
    };

    v8f acc[2][4];
    const v8f vzero = {0.f, 0.f, 0.f, 0.f, 0.f, 0.f, 0.f, 0.f};
    #pragma unroll
    for (int i = 0; i < 2; ++i)
        #pragma unroll
        for (int j = 0; j < 4; ++j) acc[i][j] = vzero;

    issue(0, 0);                                   // prologue: slab 0 in flight

    for (int kt = 0; kt < KSTEPS; ++kt) {
        const int buf = kt & 1;
        if (kt + 1 < KSTEPS) {
            issue(kt + 1, buf ^ 1);                // next slab in flight
            WAIT_ASYNC(8);                         // in-order => slab kt landed
        } else {
            WAIT_ASYNC(0);
        }
        __syncthreads();                           // all waves' portions landed

        const __bf16* pXhi = slab(buf, 0);
        const __bf16* pXlo = slab(buf, 1);
        const __bf16* pWhi = slab(buf, 2);
        const __bf16* pWlo = slab(buf, 3);

        v16bf aHi[2], aLo[2];
        #pragma unroll
        for (int tm = 0; tm < 2; ++tm) {
            const int row = wm * 32 + tm * 16 + l16;
            aHi[tm] = frag_from_lds(pXhi + row * LDSS, kh);
            aLo[tm] = frag_from_lds(pXlo + row * LDSS, kh);
        }
        #pragma unroll
        for (int tn = 0; tn < 4; ++tn) {
            const int col = wn * 64 + tn * 16 + l16;
            const v16bf bHi = frag_from_lds(pWhi + col * LDSS, kh);
            const v16bf bLo = frag_from_lds(pWlo + col * LDSS, kh);
            #pragma unroll
            for (int tm = 0; tm < 2; ++tm) {
                acc[tm][tn] = __builtin_amdgcn_wmma_f32_16x16x32_bf16(
                    false, aHi[tm], false, bHi, (short)0, acc[tm][tn], false, false);
                acc[tm][tn] = __builtin_amdgcn_wmma_f32_16x16x32_bf16(
                    false, aHi[tm], false, bLo, (short)0, acc[tm][tn], false, false);
                acc[tm][tn] = __builtin_amdgcn_wmma_f32_16x16x32_bf16(
                    false, aLo[tm], false, bHi, (short)0, acc[tm][tn], false, false);
            }
        }
        __syncthreads();                           // done reading buf
    }

    // Epilogue: out = acc * mask + bias
    // f32 C/D layout: VGPR v -> M = v + 8*(lane/16); N = lane%16.
    #pragma unroll
    for (int tm = 0; tm < 2; ++tm) {
        const int rowb = bm + wm * 32 + tm * 16 + kh * 8;
        #pragma unroll
        for (int tn = 0; tn < 4; ++tn) {
            const int col = bn + wn * 64 + tn * 16 + l16;
            const float bv = BIAS[col];
            #pragma unroll
            for (int v = 0; v < 8; ++v) {
                const size_t off = (size_t)(rowb + v) * NDIM + col;
                OUT[off] = acc[tm][tn][v] * MSK[off] + bv;
            }
        }
    }
}

// ---------------------------------------------------------------------------
// Fallback: round-1 fused kernel (used only if d_ws is too small).
// ---------------------------------------------------------------------------
__global__ __launch_bounds__(THREADS, 1)
void masked_linear_fused(const float* __restrict__ X, const float* __restrict__ MSK,
                         const float* __restrict__ W, const float* __restrict__ BIAS,
                         float* __restrict__ OUT)
{
    __shared__ __bf16 sXhi[BM][LDSS];
    __shared__ __bf16 sXlo[BM][LDSS];
    __shared__ __bf16 sWhi[BN][LDSS];
    __shared__ __bf16 sWlo[BN][LDSS];

    const int tid  = threadIdx.x;
    const int lane = tid & 31;
    const int wv   = tid >> 5;
    const int wm   = wv & 3;
    const int wn   = wv >> 2;
    const int l16  = lane & 15;
    const int kh   = lane >> 4;
    const int bm = blockIdx.y * BM;
    const int bn = blockIdx.x * BN;

    v8f acc[2][4];
    const v8f vzero = {0.f, 0.f, 0.f, 0.f, 0.f, 0.f, 0.f, 0.f};
    #pragma unroll
    for (int i = 0; i < 2; ++i)
        #pragma unroll
        for (int j = 0; j < 4; ++j) acc[i][j] = vzero;

    int r_[4], c_[4];
    #pragma unroll
    for (int i = 0; i < 4; ++i) {
        const int idx = tid + i * THREADS;
        r_[i] = idx >> 3;
        c_[i] = (idx & 7) << 2;
    }
    v4f xr[4], wr[4];
    #pragma unroll
    for (int i = 0; i < 4; ++i) {
        xr[i] = *(const v4f*)(X + (size_t)(bm + r_[i]) * KDIM + c_[i]);
        wr[i] = *(const v4f*)(W + (size_t)(bn + r_[i]) * KDIM + c_[i]);
    }

    for (int kt = 0; kt < KSTEPS; ++kt) {
        #pragma unroll
        for (int i = 0; i < 4; ++i) {
            #pragma unroll
            for (int e = 0; e < 4; ++e) {
                const float fx = xr[i][e];
                const __bf16 hx = (__bf16)fx;
                sXhi[r_[i]][c_[i] + e] = hx;
                sXlo[r_[i]][c_[i] + e] = (__bf16)(fx - (float)hx);
                const float fw = wr[i][e];
                const __bf16 hw = (__bf16)fw;
                sWhi[r_[i]][c_[i] + e] = hw;
                sWlo[r_[i]][c_[i] + e] = (__bf16)(fw - (float)hw);
            }
        }
        __syncthreads();
        if (kt + 1 < KSTEPS) {
            const int ko = (kt + 1) * BK;
            #pragma unroll
            for (int i = 0; i < 4; ++i) {
                xr[i] = *(const v4f*)(X + (size_t)(bm + r_[i]) * KDIM + ko + c_[i]);
                wr[i] = *(const v4f*)(W + (size_t)(bn + r_[i]) * KDIM + ko + c_[i]);
            }
        }
        v16bf aHi[2], aLo[2];
        #pragma unroll
        for (int tm = 0; tm < 2; ++tm) {
            const int row = wm * 32 + tm * 16 + l16;
            aHi[tm] = frag_from_lds(&sXhi[row][0], kh);
            aLo[tm] = frag_from_lds(&sXlo[row][0], kh);
        }
        #pragma unroll
        for (int tn = 0; tn < 4; ++tn) {
            const int col = wn * 64 + tn * 16 + l16;
            const v16bf bHi = frag_from_lds(&sWhi[col][0], kh);
            const v16bf bLo = frag_from_lds(&sWlo[col][0], kh);
            #pragma unroll
            for (int tm = 0; tm < 2; ++tm) {
                acc[tm][tn] = __builtin_amdgcn_wmma_f32_16x16x32_bf16(
                    false, aHi[tm], false, bHi, (short)0, acc[tm][tn], false, false);
                acc[tm][tn] = __builtin_amdgcn_wmma_f32_16x16x32_bf16(
                    false, aHi[tm], false, bLo, (short)0, acc[tm][tn], false, false);
                acc[tm][tn] = __builtin_amdgcn_wmma_f32_16x16x32_bf16(
                    false, aLo[tm], false, bHi, (short)0, acc[tm][tn], false, false);
            }
        }
        __syncthreads();
    }

    #pragma unroll
    for (int tm = 0; tm < 2; ++tm) {
        const int rowb = bm + wm * 32 + tm * 16 + kh * 8;
        #pragma unroll
        for (int tn = 0; tn < 4; ++tn) {
            const int col = bn + wn * 64 + tn * 16 + l16;
            const float bv = BIAS[col];
            #pragma unroll
            for (int v = 0; v < 8; ++v) {
                const size_t off = (size_t)(rowb + v) * NDIM + col;
                OUT[off] = acc[tm][tn][v] * MSK[off] + bv;
            }
        }
    }
}

// ---------------------------------------------------------------------------
extern "C" void kernel_launch(void* const* d_in, const int* in_sizes, int n_in,
                              void* d_out, int out_size, void* d_ws, size_t ws_size,
                              hipStream_t stream) {
    const float* x    = (const float*)d_in[0];  // (8192, 1024)
    const float* mask = (const float*)d_in[1];  // (8192, 1024)
    const float* w    = (const float*)d_in[2];  // (1024, 1024) row-major (OUT, IN)
    const float* bias = (const float*)d_in[3];  // (1024,)
    float* out = (float*)d_out;                 // (8192, 1024)

    const size_t nX = (size_t)BATCH * KDIM;     // 8M elems
    const size_t nW = (size_t)NDIM  * KDIM;     // 1M elems
    const size_t ws_needed = (2 * nX + 2 * nW) * sizeof(__bf16);  // 36MB

    dim3 grid(NDIM / BN, BATCH / BM);           // (8, 64)

    if (ws_size >= ws_needed) {
        __bf16* xhi = (__bf16*)d_ws;
        __bf16* xlo = xhi + nX;
        __bf16* whi = xlo + nX;
        __bf16* wlo = whi + nW;

        const int nsplit = (int)((nX + nW) / 4 / THREADS);   // 9216 blocks
        split_bf16<<<nsplit, THREADS, 0, stream>>>(x, w, xhi, xlo, whi, wlo);

        const size_t lds_bytes = (size_t)2 * 4 * BM * LDSS * sizeof(__bf16); // 80KB
        gemm_bf16x3_async<<<grid, THREADS, lds_bytes, stream>>>(
            xhi, xlo, whi, wlo, mask, bias, out);
    } else {
        masked_linear_fused<<<grid, THREADS, 0, stream>>>(x, mask, w, bias, out);
    }
}
